// Pointnet2MSG_10634339025564
// MI455X (gfx1250) — compile-verified, matrix operand order checked by reading, not run on hardware
//
#include <hip/hip_runtime.h>
#include <hip/hip_bf16.h>
#include <math.h>

typedef __attribute__((ext_vector_type(16))) _Float16 v16h;
typedef __attribute__((ext_vector_type(8)))  _Float16 v8h;
typedef __attribute__((ext_vector_type(8)))  float    v8f;

#define BN_EPS 1e-5f

// ---------------------------------------------------------------------------
// Extract xyz (stride-7 pointcloud -> packed stride-3)
// ---------------------------------------------------------------------------
__global__ void extract_xyz_kernel(const float* __restrict__ pc, float* __restrict__ xyz, int BN) {
    int t = blockIdx.x * blockDim.x + threadIdx.x;
    if (t >= BN) return;
    xyz[t * 3 + 0] = pc[t * 7 + 0];
    xyz[t * 3 + 1] = pc[t * 7 + 1];
    xyz[t * 3 + 2] = pc[t * 7 + 2];
}

// ---------------------------------------------------------------------------
// Farthest point sampling: one block per batch, deterministic LDS argmax.
// ---------------------------------------------------------------------------
__global__ void fps_kernel(const float* __restrict__ xyz, float* __restrict__ mind,
                           int* __restrict__ out, int N, int npoint) {
    const int T = 512;
    int b = blockIdx.x;
    int tid = threadIdx.x;
    const float* p = xyz + (size_t)b * N * 3;
    float* md = mind + (size_t)b * N;
    int* o = out + (size_t)b * npoint;
    __shared__ float sv[T];
    __shared__ int   si[T];

    for (int n = tid; n < N; n += T) md[n] = 1e10f;
    __syncthreads();

    int last = 0;
    for (int j = 0; j < npoint; ++j) {
        if (tid == 0) o[j] = last;
        float lx = p[last * 3 + 0], ly = p[last * 3 + 1], lz = p[last * 3 + 2];
        float best = -1.0f; int bi = 0;
        for (int n = tid; n < N; n += T) {
            float dx = p[n * 3 + 0] - lx;
            float dy = p[n * 3 + 1] - ly;
            float dz = p[n * 3 + 2] - lz;
            float d = fminf(md[n], dx * dx + dy * dy + dz * dz);
            md[n] = d;
            if (d > best) { best = d; bi = n; }
        }
        sv[tid] = best; si[tid] = bi;
        __syncthreads();
        for (int st = T / 2; st > 0; st >>= 1) {
            if (tid < st) {
                if (sv[tid + st] > sv[tid] ||
                    (sv[tid + st] == sv[tid] && si[tid + st] < si[tid])) {
                    sv[tid] = sv[tid + st];
                    si[tid] = si[tid + st];
                }
            }
            __syncthreads();
        }
        last = si[0];
        __syncthreads();
    }
}

// ---------------------------------------------------------------------------
// Gather sampled coordinates
// ---------------------------------------------------------------------------
__global__ void gather_xyz_kernel(const float* __restrict__ xyz, const int* __restrict__ idx,
                                  float* __restrict__ out, int B, int N, int S) {
    int t = blockIdx.x * blockDim.x + threadIdx.x;
    if (t >= B * S) return;
    int b = t / S;
    int id = idx[t];
    const float* p = xyz + ((size_t)b * N + id) * 3;
    float* o = out + (size_t)t * 3;
    o[0] = p[0]; o[1] = p[1]; o[2] = p[2];
}

// ---------------------------------------------------------------------------
// Ball query (ascending indices, pad with first hit, 0 if none)
// ---------------------------------------------------------------------------
__global__ void ball_query_kernel(float radius, int K, const float* __restrict__ xyz,
                                  const float* __restrict__ new_xyz, int* __restrict__ out,
                                  int B, int N, int S) {
    int t = blockIdx.x * blockDim.x + threadIdx.x;
    if (t >= B * S) return;
    int b = t / S;
    const float* q = new_xyz + (size_t)t * 3;
    const float* p = xyz + (size_t)b * N * 3;
    int* o = out + (size_t)t * K;
    float r2 = radius * radius;
    float qx = q[0], qy = q[1], qz = q[2];
    int cnt = 0;
    for (int n = 0; n < N && cnt < K; ++n) {
        float dx = p[n * 3 + 0] - qx;
        float dy = p[n * 3 + 1] - qy;
        float dz = p[n * 3 + 2] - qz;
        if (dx * dx + dy * dy + dz * dz < r2) o[cnt++] = n;
    }
    int first = (cnt > 0) ? o[0] : 0;
    for (int k = cnt; k < K; ++k) o[k] = first;
}

// ---------------------------------------------------------------------------
// Grouping into channel-padded f16 activation (row stride Cpad, pad zeroed).
// Feature source is either f32 (stage 0: interleaved pointcloud) or f16.
// ---------------------------------------------------------------------------
__global__ void group_kernel(const float* __restrict__ xyz,
                             const float* __restrict__ feats32,
                             const _Float16* __restrict__ feats16, int ldf,
                             const float* __restrict__ new_xyz,
                             const int* __restrict__ idx, _Float16* __restrict__ out,
                             int B, int N, int S, int K, int Cf, int Cpad) {
    int t = blockIdx.x * blockDim.x + threadIdx.x;
    int total = B * S * K;
    if (t >= total) return;
    int s = (t / K) % S;
    int b = t / (K * S);
    int id = idx[t];
    const float* p = xyz + ((size_t)b * N + id) * 3;
    const float* q = new_xyz + ((size_t)b * S + s) * 3;
    _Float16* o = out + (size_t)t * Cpad;
    o[0] = (_Float16)(p[0] - q[0]);
    o[1] = (_Float16)(p[1] - q[1]);
    o[2] = (_Float16)(p[2] - q[2]);
    if (feats32) {
        const float* f = feats32 + ((size_t)b * N + id) * ldf;
        for (int c = 0; c < Cf; ++c) o[3 + c] = (_Float16)f[c];
    } else {
        const _Float16* f = feats16 + ((size_t)b * N + id) * ldf;
        for (int c = 0; c < Cf; ++c) o[3 + c] = f[c];
    }
    for (int c = 3 + Cf; c < Cpad; ++c) o[c] = (_Float16)0.0f;
}

// ---------------------------------------------------------------------------
// Pack weights W[N,K] (f32) into f16 WMMA-B fragments, zero-padded to
// Npad x Kpad.  Layout: ((tn16*nchunk + chunk)*32 + lane)*16 + i, where the
// 16 halves of lane (h=lane>>4, l=lane&15) hold col n = tn16*16+l and
// K = chunk*32 + (i<8 ? h*8+i : 16+h*8+(i-8)).
// ---------------------------------------------------------------------------
__global__ void pack_w_kernel(const float* __restrict__ W, _Float16* __restrict__ Wpk,
                              int N, int K, int Npad, int Kpad) {
    int t = blockIdx.x * blockDim.x + threadIdx.x;
    int nchunk = Kpad >> 5;
    int total = (Npad >> 4) * nchunk * 512;
    if (t >= total) return;
    int i    = t & 15;
    int lane = (t >> 4) & 31;
    int rest = t >> 9;              // tn16*nchunk + chunk
    int chunk = rest % nchunk;
    int tn16  = rest / nchunk;
    int h = lane >> 4, l = lane & 15;
    int n  = tn16 * 16 + l;
    int kk = chunk * 32 + ((i < 8) ? (h * 8 + i) : (8 + h * 8 + i));
    float v = (n < N && kk < K) ? W[(size_t)n * K + kk] : 0.0f;
    Wpk[t] = (_Float16)v;
}

// ---------------------------------------------------------------------------
// A-fragment loader: 16 halves from a padded f16 row (two aligned 16B loads).
// ---------------------------------------------------------------------------
__device__ __forceinline__ v16h load_a_frag(const _Float16* __restrict__ arow, int k0, int h) {
    v8h lo = *(const v8h*)(arow + k0 + h * 8);
    v8h hi = *(const v8h*)(arow + k0 + 16 + h * 8);
    v16h a;
#pragma unroll
    for (int i = 0; i < 8; ++i) { a[i] = lo[i]; a[8 + i] = hi[i]; }
    return a;
}

// ---------------------------------------------------------------------------
// WMMA GEMM, 2x2 tiles per wave:  C[M,Npad](f32) = A[M,Kpad](f16) * Wpk^T.
// M % 32 == 0, Npad % 32 == 0, Kpad % 32 == 0; no predicates in the K loop.
// ---------------------------------------------------------------------------
__global__ void gemm_wmma_kernel(const _Float16* __restrict__ A, const _Float16* __restrict__ Wpk,
                                 float* __restrict__ C, const float* __restrict__ bias,
                                 int M, int Npad, int Kpad) {
    int gwave = (blockIdx.x * blockDim.x + threadIdx.x) >> 5;
    int lane  = threadIdx.x & 31;
    int tilesM = M >> 5;
    int tilesN = Npad >> 5;
    if (gwave >= tilesM * tilesN) return;
    int tm = gwave / tilesN;
    int tn = gwave % tilesN;
    int h = lane >> 4;
    int l = lane & 15;
    int nchunk = Kpad >> 5;

    const _Float16* arow0 = A + (size_t)(tm * 32 + l) * Kpad;
    const _Float16* arow1 = arow0 + (size_t)16 * Kpad;
    const v16h* wv = (const v16h*)Wpk;
    size_t wb0 = ((size_t)(tn * 2 + 0) * nchunk) * 32 + lane;
    size_t wb1 = ((size_t)(tn * 2 + 1) * nchunk) * 32 + lane;

    v8f acc00 = {}, acc01 = {}, acc10 = {}, acc11 = {};
    for (int ch = 0; ch < nchunk; ++ch) {
        int k0 = ch << 5;
        if (ch + 1 < nchunk) {
            __builtin_prefetch(arow0 + k0 + 32, 0, 1);
            __builtin_prefetch(arow1 + k0 + 32, 0, 1);
        }
        v16h a0 = load_a_frag(arow0, k0, h);
        v16h a1 = load_a_frag(arow1, k0, h);
        v16h b0 = wv[wb0 + (size_t)ch * 32];
        v16h b1 = wv[wb1 + (size_t)ch * 32];
        acc00 = __builtin_amdgcn_wmma_f32_16x16x32_f16(false, a0, false, b0, (short)0, acc00, false, false);
        acc01 = __builtin_amdgcn_wmma_f32_16x16x32_f16(false, a0, false, b1, (short)0, acc01, false, false);
        acc10 = __builtin_amdgcn_wmma_f32_16x16x32_f16(false, a1, false, b0, (short)0, acc10, false, false);
        acc11 = __builtin_amdgcn_wmma_f32_16x16x32_f16(false, a1, false, b1, (short)0, acc11, false, false);
    }

    int col0 = tn * 32 + l;
    int col1 = col0 + 16;
    float bv0 = bias ? bias[col0] : 0.0f;
    float bv1 = bias ? bias[col1] : 0.0f;
#pragma unroll
    for (int r = 0; r < 8; ++r) {
        int row0 = tm * 32 + r + h * 8;   // C/D layout: VGPR r -> row r (h=0) / r+8 (h=1)
        int row1 = row0 + 16;
        C[(size_t)row0 * Npad + col0] = acc00[r] + bv0;
        C[(size_t)row0 * Npad + col1] = acc01[r] + bv1;
        C[(size_t)row1 * Npad + col0] = acc10[r] + bv0;
        C[(size_t)row1 * Npad + col1] = acc11[r] + bv1;
    }
}

// ---------------------------------------------------------------------------
// BatchNorm statistics on f32 GEMM output: one block per channel (LDS tree).
// ---------------------------------------------------------------------------
__global__ void bn_stats_kernel(const float* __restrict__ x, float* __restrict__ sum,
                                float* __restrict__ sq, int M, int ld) {
    int c = blockIdx.x;
    int tid = threadIdx.x;
    float s = 0.0f, q = 0.0f;
    for (int m = tid; m < M; m += blockDim.x) {
        float v = x[(size_t)m * ld + c];
        s += v; q += v * v;
    }
    __shared__ float ss[256], qq[256];
    ss[tid] = s; qq[tid] = q;
    __syncthreads();
    for (int st = blockDim.x / 2; st > 0; st >>= 1) {
        if (tid < st) { ss[tid] += ss[tid + st]; qq[tid] += qq[tid + st]; }
        __syncthreads();
    }
    if (tid == 0) { sum[c] = ss[0]; sq[c] = qq[0]; }
}

// ---------------------------------------------------------------------------
// BN + ReLU: reads f32 GEMM output (stride ld), writes f16 activation
// (same stride), zeroing the pad columns c in [C, ld).
// ---------------------------------------------------------------------------
__global__ void bn_apply_relu_kernel(const float* __restrict__ x, _Float16* __restrict__ y,
                                     const float* __restrict__ sum, const float* __restrict__ sq,
                                     const float* __restrict__ gamma, const float* __restrict__ beta,
                                     int M, int C, int ld) {
    long t = (long)blockIdx.x * blockDim.x + threadIdx.x;
    long total = (long)M * ld;
    if (t >= total) return;
    int c = (int)(t % ld);
    float outv = 0.0f;
    if (c < C) {
        float inv = 1.0f / (float)M;
        float mean = sum[c] * inv;
        float var = sq[c] * inv - mean * mean;
        float v = (x[t] - mean) * rsqrtf(var + BN_EPS) * gamma[c] + beta[c];
        outv = fmaxf(v, 0.0f);
    }
    y[t] = (_Float16)outv;
}

// ---------------------------------------------------------------------------
// Max-pool over neighbor dim K (f16 in stride ld, f16 out stride ldo @ coff).
// ---------------------------------------------------------------------------
__global__ void maxpool_kernel(const _Float16* __restrict__ x, _Float16* __restrict__ out,
                               int BS, int K, int C, int ld, int ldo, int coff) {
    int t = blockIdx.x * blockDim.x + threadIdx.x;
    if (t >= BS * C) return;
    int c = t % C;
    int r = t / C;
    const _Float16* xp = x + (size_t)r * K * ld + c;
    float m = (float)xp[0];
    for (int k = 1; k < K; ++k) m = fmaxf(m, (float)xp[(size_t)k * ld]);
    out[(size_t)r * ldo + coff + c] = (_Float16)m;
}

// ---------------------------------------------------------------------------
// Head finals (h is f16 post-BN activation)
// ---------------------------------------------------------------------------
__global__ void head_score_kernel(const _Float16* __restrict__ h, const float* __restrict__ w2,
                                  const float* __restrict__ b2, float* __restrict__ out,
                                  int M, int C) {
    int m = blockIdx.x * blockDim.x + threadIdx.x;
    if (m >= M) return;
    float acc = b2[0];
    const _Float16* hr = h + (size_t)m * C;
    for (int c = 0; c < C; ++c) acc += (float)hr[c] * w2[c];
    out[m] = 1.0f / (1.0f + expf(-acc));
}

__global__ void head_reg_kernel(const _Float16* __restrict__ h2, const float* __restrict__ w2,
                                const float* __restrict__ b2, const float* __restrict__ seed,
                                float* __restrict__ out, int M, int C) {
    int m = blockIdx.x * blockDim.x + threadIdx.x;
    if (m >= M) return;
    const _Float16* hr = h2 + (size_t)m * C;
    for (int j = 0; j < 3; ++j) {
        float acc = b2[j];
        const float* wr = w2 + (size_t)j * C;
        for (int c = 0; c < C; ++c) acc += (float)hr[c] * wr[c];
        out[m * 3 + j] = acc + seed[m * 3 + j];
    }
}

// ---------------------------------------------------------------------------
// Host orchestration
// ---------------------------------------------------------------------------
static inline int cdiv(long a, long b) { return (int)((a + b - 1) / b); }
static inline int ceil32(int x) { return (x + 31) & ~31; }

extern "C" void kernel_launch(void* const* d_in, const int* in_sizes, int n_in,
                              void* d_out, int out_size, void* d_ws, size_t ws_size,
                              hipStream_t stream) {
    (void)in_sizes; (void)n_in; (void)out_size; (void)ws_size;

    const int B = 2;
    const float* pc = (const float*)d_in[0];   // [2,8192,7]

    const int mlpspec[3][2][4] = {
        {{  7,  16,  16,  32}, {  7,  32,  32,  64}},
        {{ 99,  64,  64, 128}, { 99,  64,  96, 128}},
        {{259, 128, 196, 256}, {259, 128, 196, 256}},
    };
    const int   stageS[3] = {4096, 2048, 1024};
    const float radii[3][2] = {{0.025f, 0.05f}, {0.05f, 0.1f}, {0.1f, 0.2f}};
    const int   nsamp[2] = {16, 32};

    const float* Wp[3][2][3];
    const float* Gp[3][2][3];
    const float* Bp[3][2][3];
    int ip = 2;
    for (int s = 0; s < 3; ++s)
        for (int sc = 0; sc < 2; ++sc)
            for (int l = 0; l < 3; ++l) {
                Wp[s][sc][l] = (const float*)d_in[ip++];
                Gp[s][sc][l] = (const float*)d_in[ip++];
                Bp[s][sc][l] = (const float*)d_in[ip++];
            }
    const float* w1_score = (const float*)d_in[ip++];
    const float* b1_score = (const float*)d_in[ip++];
    const float* g_score  = (const float*)d_in[ip++];
    const float* be_score = (const float*)d_in[ip++];
    const float* w2_score = (const float*)d_in[ip++];
    const float* b2_score = (const float*)d_in[ip++];
    const float* w1_reg   = (const float*)d_in[ip++];
    const float* b1_reg   = (const float*)d_in[ip++];
    const float* g_reg    = (const float*)d_in[ip++];
    const float* be_reg   = (const float*)d_in[ip++];
    const float* w2_reg   = (const float*)d_in[ip++];
    const float* b2_reg   = (const float*)d_in[ip++];

    // ---- workspace carve
    unsigned char* base = (unsigned char*)d_ws;
    size_t off = 0;
    auto alloc_f = [&](size_t n) -> float* {
        float* p = (float*)(base + off);
        off += ((n * sizeof(float) + 255) & ~(size_t)255);
        return p;
    };
    auto alloc_i = [&](size_t n) -> int* {
        int* p = (int*)(base + off);
        off += ((n * sizeof(int) + 255) & ~(size_t)255);
        return p;
    };
    auto alloc_h = [&](size_t n) -> _Float16* {
        _Float16* p = (_Float16*)(base + off);
        off += ((n * sizeof(_Float16) + 255) & ~(size_t)255);
        return p;
    };
    const size_t ACT_ELEMS = 19000000;           // covers 65536 x 288 padded
    float*    actF  = alloc_f(ACT_ELEMS);        // f32 GEMM output (pre-BN)
    _Float16* actH0 = alloc_h(ACT_ELEMS);        // f16 activations ping
    _Float16* actH1 = alloc_h(ACT_ELEMS);        // f16 activations pong
    float* mind   = alloc_f((size_t)B * 8192);
    int*   fpsidx = alloc_i((size_t)B * 4096);
    int*   ballid = alloc_i((size_t)B * 4096 * 32);
    float* xyz0   = alloc_f((size_t)B * 8192 * 3);
    float* xyzbuf[3]  = { alloc_f((size_t)B * 4096 * 3),
                          alloc_f((size_t)B * 2048 * 3),
                          alloc_f((size_t)B * 1024 * 3) };
    _Float16* featbuf[3] = { alloc_h((size_t)B * 4096 * 96),
                             alloc_h((size_t)B * 2048 * 256),
                             alloc_h((size_t)B * 1024 * 512) };
    float* bnsum = alloc_f(256);
    float* bnsq  = alloc_f(256);

    // packed f16 weights (18 MLP layers + 2 head layers)
    _Float16* WpkMLP[3][2][3];
    for (int s = 0; s < 3; ++s)
        for (int sc = 0; sc < 2; ++sc)
            for (int l = 0; l < 3; ++l) {
                int Np = ceil32(mlpspec[s][sc][l + 1]);
                int Kp = ceil32(mlpspec[s][sc][l]);
                WpkMLP[s][sc][l] = alloc_h((size_t)Np * Kp);
            }
    _Float16* WpkScore = alloc_h((size_t)256 * 512);
    _Float16* WpkReg   = alloc_h((size_t)256 * 512);

    // ---- pack all weights (cheap, L2-resident, deterministic)
    for (int s = 0; s < 3; ++s)
        for (int sc = 0; sc < 2; ++sc)
            for (int l = 0; l < 3; ++l) {
                int N  = mlpspec[s][sc][l + 1];
                int K  = mlpspec[s][sc][l];
                int Np = ceil32(N), Kp = ceil32(K);
                long tot = (long)(Np / 16) * (Kp / 32) * 512;
                pack_w_kernel<<<cdiv(tot, 256), 256, 0, stream>>>(
                    Wp[s][sc][l], WpkMLP[s][sc][l], N, K, Np, Kp);
            }
    {
        long tot = (long)(256 / 16) * (512 / 32) * 512;
        pack_w_kernel<<<cdiv(tot, 256), 256, 0, stream>>>(w1_score, WpkScore, 256, 512, 256, 512);
        pack_w_kernel<<<cdiv(tot, 256), 256, 0, stream>>>(w1_reg,   WpkReg,   256, 512, 256, 512);
    }

    // ---- stage 0 xyz extraction
    extract_xyz_kernel<<<cdiv((long)B * 8192, 256), 256, 0, stream>>>(pc, xyz0, B * 8192);

    const float*    cur_xyz    = xyz0;
    const float*    cur_feat32 = pc + 3;   // stage 0: interleaved f32, ldf = 7
    const _Float16* cur_feat16 = nullptr;
    int cur_ldf = 7;
    int cur_Cf  = 4;
    int N = 8192;

    for (int s = 0; s < 3; ++s) {
        int S = stageS[s];
        float* new_xyz = xyzbuf[s];

        fps_kernel<<<B, 512, 0, stream>>>(cur_xyz, mind, fpsidx, N, S);
        gather_xyz_kernel<<<cdiv((long)B * S, 256), 256, 0, stream>>>(cur_xyz, fpsidx, new_xyz, B, N, S);

        int Cout_total = mlpspec[s][0][3] + mlpspec[s][1][3];
        _Float16* outfeat = featbuf[s];
        int coff = 0;

        for (int sc = 0; sc < 2; ++sc) {
            int K = nsamp[sc];
            float rad = radii[s][sc];
            ball_query_kernel<<<cdiv((long)B * S, 256), 256, 0, stream>>>(
                rad, K, cur_xyz, new_xyz, ballid, B, N, S);

            int M = B * S * K;                     // multiple of 32
            int Cpad0 = ceil32(mlpspec[s][sc][0]); // padded input channels
            group_kernel<<<cdiv((long)M, 256), 256, 0, stream>>>(
                cur_xyz, cur_feat32, cur_feat16, cur_ldf, new_xyz, ballid,
                actH0, B, N, S, K, cur_Cf, Cpad0);

            _Float16* in  = actH0;
            _Float16* outh = actH1;
            for (int l = 0; l < 3; ++l) {
                int cin  = mlpspec[s][sc][l];
                int cout = mlpspec[s][sc][l + 1];
                int Kp = ceil32(cin), Np = ceil32(cout);
                long waves = (long)(M / 32) * (Np / 32);
                gemm_wmma_kernel<<<cdiv(waves * 32, 256), 256, 0, stream>>>(
                    in, WpkMLP[s][sc][l], actF, nullptr, M, Np, Kp);
                bn_stats_kernel<<<cout, 256, 0, stream>>>(actF, bnsum, bnsq, M, Np);
                bn_apply_relu_kernel<<<cdiv((long)M * Np, 256), 256, 0, stream>>>(
                    actF, outh, bnsum, bnsq, Gp[s][sc][l], Bp[s][sc][l], M, cout, Np);
                _Float16* t = in; in = outh; outh = t;
            }
            int clast = mlpspec[s][sc][3];
            int ldlast = ceil32(clast);
            maxpool_kernel<<<cdiv((long)B * S * clast, 256), 256, 0, stream>>>(
                in, outfeat, B * S, K, clast, ldlast, Cout_total, coff);
            coff += clast;
        }

        cur_xyz    = new_xyz;
        cur_feat32 = nullptr;
        cur_feat16 = outfeat;
        cur_ldf    = Cout_total;
        cur_Cf     = Cout_total;
        N = S;
    }

    // ---- heads: flat = featbuf[2]  [2048, 512] f16 (512 % 32 == 0)
    const int HM = B * 1024;
    const int HC = 512;
    const int HH = 256;
    _Float16* flat  = featbuf[2];
    _Float16* hbuf  = actH0;
    _Float16* h2buf = actH1;
    float* out_reg   = (float*)d_out;
    float* out_score = (float*)d_out + (size_t)HM * 3;
    float* out_seed  = out_score + HM;

    {
        long waves = (long)(HM / 32) * (HH / 32);
        gemm_wmma_kernel<<<cdiv(waves * 32, 256), 256, 0, stream>>>(
            flat, WpkScore, actF, b1_score, HM, HH, HC);
        bn_stats_kernel<<<HH, 256, 0, stream>>>(actF, bnsum, bnsq, HM, HH);
        bn_apply_relu_kernel<<<cdiv((long)HM * HH, 256), 256, 0, stream>>>(
            actF, hbuf, bnsum, bnsq, g_score, be_score, HM, HH, HH);
        head_score_kernel<<<cdiv(HM, 256), 256, 0, stream>>>(
            hbuf, w2_score, b2_score, out_score, HM, HH);
    }
    {
        long waves = (long)(HM / 32) * (HH / 32);
        gemm_wmma_kernel<<<cdiv(waves * 32, 256), 256, 0, stream>>>(
            flat, WpkReg, actF, b1_reg, HM, HH, HC);
        bn_stats_kernel<<<HH, 256, 0, stream>>>(actF, bnsum, bnsq, HM, HH);
        bn_apply_relu_kernel<<<cdiv((long)HM * HH, 256), 256, 0, stream>>>(
            actF, h2buf, bnsum, bnsq, g_reg, be_reg, HM, HH, HH);
        head_reg_kernel<<<cdiv(HM, 256), 256, 0, stream>>>(
            h2buf, w2_reg, b2_reg, xyzbuf[2], out_reg, HM, HH);
    }
    hipMemcpyAsync(out_seed, xyzbuf[2], (size_t)HM * 3 * sizeof(float),
                   hipMemcpyDeviceToDevice, stream);
}
